// Attention_48627619726070
// MI455X (gfx1250) — compile-verified, matrix operand order checked by reading, not run on hardware
//
#include <hip/hip_runtime.h>

// MI455X / gfx1250, wave32. Flash attention: v_wmma_f32_16x16x32_bf16 compute,
// Tensor Data Mover (tensor_load_to_lds) double-buffered K / V^T staging.

typedef unsigned int uint;
typedef __attribute__((ext_vector_type(16))) __bf16 v16bf;
typedef __attribute__((ext_vector_type(8)))  __bf16 v8bf;
typedef __attribute__((ext_vector_type(8)))  float  v8f;
typedef __attribute__((ext_vector_type(4)))  uint   v4u;
typedef __attribute__((ext_vector_type(8)))  uint   v8u;

// 16-bit A/B fragment, row-major source: lane holds row `row`, element e maps to
// k = k0 + (e>=8?16:0) + half*8 + (e&7)   (CDNA5 ISA 7.12.2, 16-bit 16x32 layout)
__device__ __forceinline__ v16bf frag_row(const __bf16* base, int stride, int row,
                                          int k0, int half) {
  const v8bf lo = *reinterpret_cast<const v8bf*>(base + row * stride + k0 + half * 8);
  const v8bf hi = *reinterpret_cast<const v8bf*>(base + row * stride + k0 + half * 8 + 16);
  v16bf f;
#pragma unroll
  for (int i = 0; i < 8; ++i) { f[i] = lo[i]; f[i + 8] = hi[i]; }
  return f;
}

// TDM: DMA a 2D bf16 tile from global into LDS.
// D# bit layout per CDNA5 ISA ch.8 (group0 128b, group1 256b), 2-group 2D form.
__device__ __forceinline__ void tdm_load_tile(const __bf16* gsrc, uint lds_off,
                                              uint dim0, uint dim1,
                                              uint tile0, uint tile1, uint stride0) {
  unsigned long long ga = (unsigned long long)(size_t)gsrc;
  v4u g0;
  g0[0] = 1u;                                  // count=1: one valid descriptor
  g0[1] = lds_off;                             // LDS byte address of tile start
  g0[2] = (uint)ga;                            // global_addr[31:0]
  g0[3] = (uint)(ga >> 32) | (2u << 30);       // global_addr[56:32] | type=2
  v8u g1;
  g1[0] = 0x00010000u;                         // workgroup_mask=0, data_size=1 (2B)
  g1[1] = dim0 << 16;                          // tensor_dim0 @ bit48
  g1[2] = (dim0 >> 16) | (dim1 << 16);         // tensor_dim1 @ bit80
  g1[3] = (dim1 >> 16) | (tile0 << 16);        // tile_dim0   @ bit112
  g1[4] = tile1;                               // tile_dim1 (tile_dim2 = 0)
  g1[5] = stride0;                             // tensor_dim0_stride[31:0]
  g1[6] = 0u;
  g1[7] = 0u;
  asm volatile("tensor_load_to_lds %0, %1" :: "s"(g0), "s"(g1) : "memory");
}

__global__ void cast_x(const float* __restrict__ x, __bf16* __restrict__ xb, int n) {
  int i = (blockIdx.x * blockDim.x + threadIdx.x) * 4;
  if (i < n) {
    const float4 v = *reinterpret_cast<const float4*>(x + i);
    xb[i]     = (__bf16)v.x;
    xb[i + 1] = (__bf16)v.y;
    xb[i + 2] = (__bf16)v.z;
    xb[i + 3] = (__bf16)v.w;
  }
}

// Wt[w][n][k] = W_w[k][n], bf16.  3 * 128 * 1024 elements.
__global__ void wtrans(const float* __restrict__ Wq, const float* __restrict__ Wk,
                       const float* __restrict__ Wv, __bf16* __restrict__ wt) {
  int idx = blockIdx.x * blockDim.x + threadIdx.x;   // 0 .. 393215
  int w   = idx >> 17;
  int rem = idx & 131071;
  int nn  = rem >> 10;
  int kk  = rem & 1023;
  const float* W = (w == 0) ? Wq : ((w == 1) ? Wk : Wv);
  wt[idx] = (__bf16)W[kk * 128 + nn];
}

// One 16x64 output slab per wave (A fragment reused across 4 N-tiles).
// Q and K stored row-major [8192][128]; V stored TRANSPOSED [128][8192] so the
// attention kernel's V B-fragments read contiguously.
__global__ __launch_bounds__(256) void qkv_gemm(
    const __bf16* __restrict__ xb, const __bf16* __restrict__ wt,
    const float* __restrict__ bq, const float* __restrict__ bk,
    const float* __restrict__ bv, __bf16* __restrict__ qb,
    __bf16* __restrict__ kb, __bf16* __restrict__ vt) {
  const int wave = threadIdx.x >> 5, lane = threadIdx.x & 31;
  const int half = lane >> 4, n16 = lane & 15;
  const int tile = blockIdx.x * 8 + wave;            // 3072 wave-slabs total
  const int tile_m = tile / 6;                       // 0..511
  const int rem = tile % 6;
  const int w = rem >> 1, ng = rem & 1;              // cols [ng*64, ng*64+64)
  const __bf16* wts = wt + w * (128 * 1024);

  v8f acc[4] = {};
  for (int kc = 0; kc < 32; ++kc) {
    v16bf a = frag_row(xb, 1024, tile_m * 16 + n16, kc * 32, half);
#pragma unroll
    for (int t = 0; t < 4; ++t) {
      v16bf b = frag_row(wts, 1024, (ng * 4 + t) * 16 + n16, kc * 32, half);
      acc[t] = __builtin_amdgcn_wmma_f32_16x16x32_bf16(false, a, false, b,
                                                       (short)0, acc[t], false, false);
    }
  }

  const float* bias = (w == 0) ? bq : ((w == 1) ? bk : bv);
  const float scale = (w == 0) ? 0.088388347648318447f : 1.0f;  // fold 1/sqrt(128) into Q
#pragma unroll
  for (int t = 0; t < 4; ++t) {
    const int col = (ng * 4 + t) * 16 + n16;
    const float bcol = bias[col];
#pragma unroll
    for (int r = 0; r < 8; ++r) {
      const int row = tile_m * 16 + half * 8 + r;    // C/D layout: m = r + 8*half
      const float val = (acc[t][r] + bcol) * scale;
      if (w == 2) vt[col * 8192 + row] = (__bf16)val;            // V transposed
      else ((w == 0) ? qb : kb)[row * 128 + col] = (__bf16)val;  // Q / K row-major
    }
  }
}

// Flash attention: 8 waves/WG * 16 query rows; TDM double-buffers a 32x128 K
// tile and a 128x32 V^T tile in LDS while all waves run WMMA + online softmax.
__global__ __launch_bounds__(256) void flash_attn(
    const __bf16* __restrict__ qb, const __bf16* __restrict__ kb,
    const __bf16* __restrict__ vt, float* __restrict__ out) {
  __shared__ __align__(16) __bf16 kT[2][32 * 128];   // K tile, row-major
  __shared__ __align__(16) __bf16 vtT[2][128 * 32];  // V^T tile: [d][token]
  __shared__ __align__(16) __bf16 pT[8][16 * 32];    // per-wave P transpose buffer

  const int wave = threadIdx.x >> 5, lane = threadIdx.x & 31;
  const int half = lane >> 4, n16 = lane & 15;
  const int q0 = blockIdx.x * 128 + wave * 16;

  v16bf aq[4];
#pragma unroll
  for (int kc = 0; kc < 4; ++kc)
    aq[kc] = frag_row(qb, 128, q0 + n16, kc * 32, half);

  v8f o[8] = {};
  float mrow[8], lrow[8];
#pragma unroll
  for (int r = 0; r < 8; ++r) { mrow[r] = -3.0e38f; lrow[r] = 0.f; }

  // Prime buffer 0 via TDM: wave 0 -> K tile, wave 1 -> V^T tile.
  if (wave == 0 && lane == 0)
    tdm_load_tile(kb, (uint)(size_t)&kT[0][0], 128u, 8192u, 128u, 32u, 128u);
  if (wave == 1 && lane == 0)
    tdm_load_tile(vt, (uint)(size_t)&vtT[0][0], 8192u, 128u, 32u, 128u, 8192u);
  if (wave < 2) __builtin_amdgcn_s_wait_tensorcnt(0);
  __syncthreads();

  int buf = 0;
  for (int j0 = 0; j0 < 8192; j0 += 32) {
    // Kick off DMA for the next tiles into the back buffer (overlaps compute).
    if ((j0 + 32) < 8192 && lane == 0) {
      if (wave == 0)
        tdm_load_tile(kb + (j0 + 32) * 128, (uint)(size_t)&kT[buf ^ 1][0],
                      128u, 8192u - (j0 + 32), 128u, 32u, 128u);
      if (wave == 1)
        tdm_load_tile(vt + (j0 + 32), (uint)(size_t)&vtT[buf ^ 1][0],
                      8192u - (j0 + 32), 128u, 32u, 128u, 8192u);
    }
    const __bf16* kcur = &kT[buf][0];
    const __bf16* vcur = &vtT[buf][0];

    // S = Q * K^T : two 16x16 tiles over 32 k-columns
    v8f s0 = {}, s1 = {};
#pragma unroll
    for (int kc = 0; kc < 4; ++kc) {
      v16bf b0 = frag_row(kcur, 128, n16, kc * 32, half);
      v16bf b1 = frag_row(kcur, 128, 16 + n16, kc * 32, half);
      s0 = __builtin_amdgcn_wmma_f32_16x16x32_bf16(false, aq[kc], false, b0,
                                                   (short)0, s0, false, false);
      s1 = __builtin_amdgcn_wmma_f32_16x16x32_bf16(false, aq[kc], false, b1,
                                                   (short)0, s1, false, false);
    }

    // Online softmax over 32 columns; rows live across 16-lane halves.
#pragma unroll
    for (int r = 0; r < 8; ++r) {
      float mx = fmaxf(s0[r], s1[r]);
#pragma unroll
      for (int off = 8; off > 0; off >>= 1)
        mx = fmaxf(mx, __shfl_xor(mx, off, 16));
      float newm = fmaxf(mrow[r], mx);
      float corr = __expf(mrow[r] - newm);
      float p0 = __expf(s0[r] - newm);
      float p1 = __expf(s1[r] - newm);
      float rs = p0 + p1;
#pragma unroll
      for (int off = 8; off > 0; off >>= 1)
        rs += __shfl_xor(rs, off, 16);
      lrow[r] = lrow[r] * corr + rs;
      mrow[r] = newm;
#pragma unroll
      for (int t = 0; t < 8; ++t) o[t][r] *= corr;
      // D-layout -> row-major 16x32 P tile in LDS (per-wave, DS ops in-order)
      pT[wave][(half * 8 + r) * 32 + n16]      = (__bf16)p0;
      pT[wave][(half * 8 + r) * 32 + 16 + n16] = (__bf16)p1;
    }

    // O += P * V : V^T tile gives contiguous B-fragments (2x b128 per frag).
    v16bf ap = frag_row(&pT[wave][0], 32, n16, 0, half);
#pragma unroll
    for (int t = 0; t < 8; ++t) {
      v16bf bvf = frag_row(vcur, 32, t * 16 + n16, 0, half);
      o[t] = __builtin_amdgcn_wmma_f32_16x16x32_bf16(false, ap, false, bvf,
                                                     (short)0, o[t], false, false);
    }

    // Issuing waves wait for the back-buffer DMA; barrier releases everyone.
    if (wave < 2) __builtin_amdgcn_s_wait_tensorcnt(0);
    __syncthreads();
    buf ^= 1;
  }

#pragma unroll
  for (int t = 0; t < 8; ++t)
#pragma unroll
    for (int r = 0; r < 8; ++r) {
      int row = q0 + half * 8 + r;
      out[row * 128 + t * 16 + n16] = o[t][r] / lrow[r];
    }
}

extern "C" void kernel_launch(void* const* d_in, const int* in_sizes, int n_in,
                              void* d_out, int out_size, void* d_ws, size_t ws_size,
                              hipStream_t stream) {
  const float* x  = (const float*)d_in[0];
  const float* Wq = (const float*)d_in[1];
  const float* bq = (const float*)d_in[2];
  const float* Wk = (const float*)d_in[3];
  const float* bk = (const float*)d_in[4];
  const float* Wv = (const float*)d_in[5];
  const float* bv = (const float*)d_in[6];
  float* out = (float*)d_out;

  char* ws = (char*)d_ws;
  __bf16* xb = (__bf16*)ws;                               // 8192*1024 bf16 = 16 MB
  __bf16* wt = (__bf16*)(ws + 16777216);                  // 3*128*1024 bf16
  __bf16* qb = (__bf16*)(ws + 16777216 + 786432);         // 8192*128 bf16
  __bf16* kb = qb + 8192 * 128;                           // 8192*128 bf16
  __bf16* vt = kb + 8192 * 128;                           // 128*8192 bf16 (transposed)

  cast_x<<<8192, 256, 0, stream>>>(x, xb, 8192 * 1024);
  wtrans<<<1536, 256, 0, stream>>>(Wq, Wk, Wv, wt);
  qkv_gemm<<<384, 256, 0, stream>>>(xb, wt, bq, bk, bv, qb, kb, vt);
  flash_attn<<<64, 256, 0, stream>>>(qb, kb, vt, out);
}
